// SchemaBert_5274219839807
// MI455X (gfx1250) — matrix operand order repaired and checked by reading, not compiled, not run on hardware
//
#include <hip/hip_runtime.h>
#include <math.h>

#define DM    1024
#define DFF   4096
#define NH    16
#define DH    64
#define LQ    256
#define NT    40
#define NC    12
#define NP    64
#define NFK   6
#define NLAY  5
#define NB    32
#define SM    (LQ + 3*NT)     /* 376 */
#define LNEPS 1e-12f

typedef __bf16 bf16_t;
typedef bf16_t v16bf  __attribute__((ext_vector_type(16)));
typedef bf16_t bf16x8 __attribute__((ext_vector_type(8)));
typedef bf16_t bf16x2 __attribute__((ext_vector_type(2)));
typedef float  v8f    __attribute__((ext_vector_type(8)));

// ---------------------------------------------------------------- helpers

__device__ __forceinline__ v8f wmma_bf16(v16bf a, v16bf b, v8f c) {
  return __builtin_amdgcn_wmma_f32_16x16x32_bf16(false, a, false, b, (short)0, c,
                                                 false, false);
}

__device__ __forceinline__ v16bf cat8(bf16x8 lo, bf16x8 hi) {
  return __builtin_shufflevector(lo, hi, 0, 1, 2, 3, 4, 5, 6, 7, 8, 9, 10, 11,
                                 12, 13, 14, 15);
}

// A-matrix fragment (16x32 bf16): lanes 0-15 hold K{0..7,16..23}, lanes 16-31
// hold K{8..15,24..31}; rowp points at LDS row (k-contiguous) for M=lane%16.
__device__ __forceinline__ v16bf load_afrag(const bf16_t* rowp, int hl) {
  const bf16_t* p = rowp + hl * 8;
  return cat8(*(const bf16x8*)p, *(const bf16x8*)(p + 16));
}
// B-matrix fragment (32x16 bf16): lanes 0-15 hold K{0..7,16..23}, lanes 16-31
// K{8..15,24..31} for column N=lane%16; colp points at LDS row (k-contiguous).
__device__ __forceinline__ v16bf load_bfrag(const bf16_t* colp, int hl) {
  const bf16_t* p = colp + hl * 8;
  return cat8(*(const bf16x8*)p, *(const bf16x8*)(p + 16));
}

// convert+stage 16 contiguous fp32 -> 16 contiguous bf16 in LDS (16B aligned)
__device__ __forceinline__ void stage_row16(bf16_t* d, const float* p) {
  const float4* p4 = (const float4*)p;
  float4 f0 = p4[0], f1 = p4[1], f2 = p4[2], f3 = p4[3];
  bf16x8 a, b;
  a[0] = (bf16_t)f0.x; a[1] = (bf16_t)f0.y; a[2] = (bf16_t)f0.z; a[3] = (bf16_t)f0.w;
  a[4] = (bf16_t)f1.x; a[5] = (bf16_t)f1.y; a[6] = (bf16_t)f1.z; a[7] = (bf16_t)f1.w;
  b[0] = (bf16_t)f2.x; b[1] = (bf16_t)f2.y; b[2] = (bf16_t)f2.z; b[3] = (bf16_t)f2.w;
  b[4] = (bf16_t)f3.x; b[5] = (bf16_t)f3.y; b[6] = (bf16_t)f3.z; b[7] = (bf16_t)f3.w;
  *(bf16x8*)d = a;
  *(bf16x8*)(d + 8) = b;
}
__device__ __forceinline__ void stage_zero16(bf16_t* d) {
  bf16x8 z = {};
  *(bf16x8*)d = z;
  *(bf16x8*)(d + 8) = z;
}

// async cache->LDS 16B copy (CDNA5 async path, ASYNCcnt tracked)
__device__ __forceinline__ void async_copy_b128(unsigned lds_off,
                                                const bf16_t* gp) {
  unsigned long long ga = (unsigned long long)(size_t)gp;
  asm volatile("global_load_async_to_lds_b128 %0, %1, off"
               :
               : "v"(lds_off), "v"(ga)
               : "memory");
}
__device__ __forceinline__ void wait_async0() {
  asm volatile("s_wait_asynccnt 0" ::: "memory");
}

__device__ __forceinline__ void block_reduce2(float& a, float& b) {
  __shared__ float ra[256];
  __shared__ float rb[256];
  int t = threadIdx.x;
  ra[t] = a; rb[t] = b;
  __syncthreads();
  for (int off = 128; off; off >>= 1) {
    if (t < off) { ra[t] += ra[t + off]; rb[t] += rb[t + off]; }
    __syncthreads();
  }
  a = ra[0]; b = rb[0];
  __syncthreads();
}

// ------------------------------------------- weight convert+transpose (once)
// W [K,N] fp32 -> WT [N,K] bf16, batched over blockIdx.z matrices.

__global__ __launch_bounds__(256) void transpose_cvt_kernel(
    const float* __restrict__ W, bf16_t* __restrict__ WT, int K, int N) {
  __shared__ float s[32][33];
  const size_t msz = (size_t)K * N;
  const float* src = W + (size_t)blockIdx.z * msz;
  bf16_t* dst = WT + (size_t)blockIdx.z * msz;
  int k0 = blockIdx.y * 32, n0 = blockIdx.x * 32;
  int tx = threadIdx.x & 31, ty = threadIdx.x >> 5;  // 32 x 8
#pragma unroll
  for (int i = 0; i < 4; ++i)
    s[ty + 8 * i][tx] = src[(size_t)(k0 + ty + 8 * i) * N + n0 + tx];
  __syncthreads();
#pragma unroll
  for (int i = 0; i < 4; ++i)
    dst[(size_t)(n0 + ty + 8 * i) * K + k0 + tx] = (bf16_t)s[tx][ty + 8 * i];
}

// ---------------------------------------------------------------- embeddings

__global__ __launch_bounds__(256) void embed_ln_kernel(
    const int* __restrict__ ids, const int* __restrict__ tids,
    const float* __restrict__ word_emb, const float* __restrict__ pos_emb,
    const float* __restrict__ type_emb, const float* __restrict__ gln,
    float* __restrict__ out, int S, int out_seq_stride) {
  int token = blockIdx.x;
  int n = token / S, s = token - n * S;
  int id = ids[token];
  int tt = tids ? tids[token] : 0;
  int c = threadIdx.x * 4;
  float4 w = *(const float4*)(word_emb + (size_t)id * DM + c);
  float4 p = *(const float4*)(pos_emb + (size_t)s * DM + c);
  float4 ty = *(const float4*)(type_emb + (size_t)tt * DM + c);
  float e[4] = {w.x + p.x + ty.x, w.y + p.y + ty.y, w.z + p.z + ty.z,
                w.w + p.w + ty.w};
  float ssum = 0.f, ssq = 0.f;
#pragma unroll
  for (int j = 0; j < 4; ++j) { ssum += e[j]; ssq += e[j] * e[j]; }
  block_reduce2(ssum, ssq);
  float mean = ssum * (1.f / DM);
  float inv = rsqrtf(ssq * (1.f / DM) - mean * mean + LNEPS);
  size_t row = (size_t)n * out_seq_stride + s;
#pragma unroll
  for (int j = 0; j < 4; ++j)
    out[row * DM + c + j] = (e[j] - mean) * inv * gln[c + j] + gln[DM + c + j];
}

__global__ __launch_bounds__(256) void addposln_kernel(
    const float* __restrict__ in, const float* __restrict__ pos_emb,
    const float* __restrict__ type_emb, const float* __restrict__ gln,
    float* __restrict__ out) {
  int b = blockIdx.x / (3 * NT);
  int i = blockIdx.x - b * (3 * NT);
  int c = threadIdx.x * 4;
  float4 x = *(const float4*)(in + ((size_t)b * (3 * NT) + i) * DM + c);
  float4 p = *(const float4*)(pos_emb + (size_t)(LQ + i) * DM + c);
  float4 ty = *(const float4*)(type_emb + (size_t)DM + c);
  float e[4] = {x.x + p.x + ty.x, x.y + p.y + ty.y, x.z + p.z + ty.z,
                x.w + p.w + ty.w};
  float ssum = 0.f, ssq = 0.f;
#pragma unroll
  for (int j = 0; j < 4; ++j) { ssum += e[j]; ssq += e[j] * e[j]; }
  block_reduce2(ssum, ssq);
  float mean = ssum * (1.f / DM);
  float inv = rsqrtf(ssq * (1.f / DM) - mean * mean + LNEPS);
  size_t row = (size_t)b * SM + LQ + i;
#pragma unroll
  for (int j = 0; j < 4; ++j)
    out[row * DM + c + j] = (e[j] - mean) * inv * gln[c + j] + gln[DM + c + j];
}

__global__ __launch_bounds__(256) void ln_kernel(const float* __restrict__ in,
                                                 const float* __restrict__ gamma,
                                                 const float* __restrict__ beta,
                                                 float* __restrict__ out) {
  size_t row = blockIdx.x;
  int c = threadIdx.x * 4;
  float4 x = *(const float4*)(in + row * DM + c);
  float e[4] = {x.x, x.y, x.z, x.w};
  float ssum = 0.f, ssq = 0.f;
#pragma unroll
  for (int j = 0; j < 4; ++j) { ssum += e[j]; ssq += e[j] * e[j]; }
  block_reduce2(ssum, ssq);
  float mean = ssum * (1.f / DM);
  float inv = rsqrtf(ssq * (1.f / DM) - mean * mean + LNEPS);
#pragma unroll
  for (int j = 0; j < 4; ++j)
    out[row * DM + c + j] = (e[j] - mean) * inv * gamma[c + j] + beta[c + j];
}

// ---------------------------------------------------------------- WMMA GEMM
// C[M,N] = epi(A[M,K] @ B[K,N] + bias + resid); B given pre-transposed bf16
// BT[N,K].  M%128==0, N%128==0, K%32==0.  256 threads = 8 waves; block tile
// 128x128x32; wave tile 32x64 (2x4 frags).  Double-buffered LDS: tile t+1's
// async B copy + A stage are issued before tile t's WMMAs, and the ASYNCcnt
// wait happens after them, so the cache->LDS copy overlaps the matrix pipe.

template <int EPI>  // 0 = none, 1 = exact gelu
__global__ __launch_bounds__(256) void gemm_bf16_kernel(
    const float* __restrict__ A, const bf16_t* __restrict__ BT,
    const float* __restrict__ bias, const float* __restrict__ resid,
    float* __restrict__ C, int M, int N, int K) {
  constexpr int LDA = 40, LDB = 40;  // bf16 row strides (8-elem pad, 16B align)
  __shared__ __align__(16) bf16_t sA[2][128 * LDA];  // [m][k]
  __shared__ __align__(16) bf16_t sB[2][128 * LDB];  // [n][k]
  const int tid = threadIdx.x;
  const int lane = tid & 31, wave = tid >> 5;
  const int wm = wave & 3;   // 32-row group
  const int wn = wave >> 2;  // 64-col group
  const int m0 = blockIdx.y * 128, n0 = blockIdx.x * 128;
  const int hl = lane >> 4, l16 = lane & 15;

  v8f acc[2][4] = {};

  const int arow = tid >> 1, acol = (tid & 1) * 16;
  const float* aptr = A + (size_t)(m0 + arow) * K + acol;
  // B tile = 128 rows x 32 bf16 = 512 x 16B chunks; 2 per thread
  const int ch0 = tid * 2;
  const int bn0 = ch0 >> 2, bj0 = ch0 & 3;      // chunk0: row, 8-elem group
  const int bn1 = (ch0 + 1) >> 2, bj1 = (ch0 + 1) & 3;
  const bf16_t* bp0 = BT + (size_t)(n0 + bn0) * K + bj0 * 8;
  const bf16_t* bp1 = BT + (size_t)(n0 + bn1) * K + bj1 * 8;
  const unsigned sBoff[2] = {(unsigned)(size_t)&sB[0][0],
                             (unsigned)(size_t)&sB[1][0]};
  const unsigned loff0 = (unsigned)(bn0 * LDB + bj0 * 8) * 2u;
  const unsigned loff1 = (unsigned)(bn1 * LDB + bj1 * 8) * 2u;
  const int ntiles = K >> 5;

  // preload tile 0 into buffer 0
  async_copy_b128(sBoff[0] + loff0, bp0);
  async_copy_b128(sBoff[0] + loff1, bp1);
  stage_row16(&sA[0][arow * LDA + acol], aptr);
  wait_async0();
  __syncthreads();

  for (int t = 0; t < ntiles; ++t) {
    const int cur = t & 1, nxt = cur ^ 1;
    if (t + 1 < ntiles) {  // prefetch next tile into the other buffer
      const int k1 = (t + 1) * 32;
      async_copy_b128(sBoff[nxt] + loff0, bp0 + k1);
      async_copy_b128(sBoff[nxt] + loff1, bp1 + k1);
      stage_row16(&sA[nxt][arow * LDA + acol], aptr + k1);
      if (t + 2 < ntiles) __builtin_prefetch(aptr + (t + 2) * 32, 0, 0);
    }

    v16bf afr[2], bfr[4];
#pragma unroll
    for (int mi = 0; mi < 2; ++mi)
      afr[mi] = load_afrag(&sA[cur][(wm * 32 + mi * 16 + l16) * LDA], hl);
#pragma unroll
    for (int ni = 0; ni < 4; ++ni)
      bfr[ni] = load_bfrag(&sB[cur][(wn * 64 + ni * 16 + l16) * LDB], hl);
#pragma unroll
    for (int mi = 0; mi < 2; ++mi)
#pragma unroll
      for (int ni = 0; ni < 4; ++ni)
        acc[mi][ni] = wmma_bf16(afr[mi], bfr[ni], acc[mi][ni]);

    if (t + 1 < ntiles) wait_async0();  // next tile landed during WMMAs
    __syncthreads();
  }

  // epilogue: C layout per frag: lane=col n, VGPR v -> row v (+8 for lanes>=16)
#pragma unroll
  for (int mi = 0; mi < 2; ++mi) {
    int rbase = m0 + wm * 32 + mi * 16 + hl * 8;
#pragma unroll
    for (int ni = 0; ni < 4; ++ni) {
      int col = n0 + wn * 64 + ni * 16 + l16;
      float bv = bias ? bias[col] : 0.f;
#pragma unroll
      for (int v = 0; v < 8; ++v) {
        size_t idx = (size_t)(rbase + v) * N + col;
        float x = acc[mi][ni][v] + bv;
        if (resid) x += resid[idx];
        if (EPI == 1) x = 0.5f * x * (1.f + erff(x * 0.70710678118654752f));
        C[idx] = x;
      }
    }
  }
}

// ----------------------------------------------------- fused flash attention
// grid.x = Batt*NH, grid.y = ceil(S/64).  q/k/v/out rows: [(b*S+s)*DM + h*64].

__global__ __launch_bounds__(256) void attn_kernel(const float* __restrict__ Qm,
                                                   const float* __restrict__ Km,
                                                   const float* __restrict__ Vm,
                                                   float* __restrict__ Om,
                                                   int S) {
  constexpr int LT = 72;  // bf16 row stride for 64-wide tiles (16B aligned)
  constexpr int LS = 68;  // f32 row stride for score tile
  __shared__ __align__(16) bf16_t sQ[64 * LT];   // [q][dh]
  __shared__ __align__(16) bf16_t sK[64 * LT];   // [key][dh]
  __shared__ __align__(16) bf16_t sVt[64 * LT];  // [dh][key]
  __shared__ __align__(16) bf16_t sP[64 * LT];   // [q][key]
  __shared__ float sS[64 * LS];
  __shared__ float sMx[64], sL[64], sAl[64], red[64 * 4];

  const int tid = threadIdx.x, lane = tid & 31, wave = tid >> 5;
  const int wm = wave & 3;    // query 16-row tile
  const int wn2 = wave >> 2;  // dh 32-col half
  const int bh = blockIdx.x;
  const int b = bh / NH, h = bh - b * NH;
  const int q0 = blockIdx.y * 64;
  const int hl = lane >> 4, l16 = lane & 15;

  {  // load Q tile (zero-padded)
    int row = tid >> 2, c16 = (tid & 3) * 16;
    bf16_t* d = sQ + row * LT + c16;
    int q = q0 + row;
    if (q < S)
      stage_row16(d, Qm + (size_t)(b * S + q) * DM + h * DH + c16);
    else
      stage_zero16(d);
  }
  if (tid < 64) { sMx[tid] = -1e30f; sL[tid] = 0.f; }

  v8f oacc[2] = {};
  const int nk = (S + 63) / 64;
  for (int kt = 0; kt < nk; ++kt) {
    const int k0t = kt * 64;
    {  // stage K tile [key][dh]
      int row = tid >> 2, c16 = (tid & 3) * 16;
      bf16_t* d = sK + row * LT + c16;
      int kk = k0t + row;
      if (kk < S)
        stage_row16(d, Km + (size_t)(b * S + kk) * DM + h * DH + c16);
      else
        stage_zero16(d);
    }
    {  // stage V transposed [dh][key], packing key-pairs as dwords
      int kp = tid >> 3, dh8 = (tid & 7) * 8;
      int kk0 = k0t + 2 * kp, kk1 = kk0 + 1;
      float r0[8] = {}, r1[8] = {};
      if (kk0 < S) {
        const float* p = Vm + (size_t)(b * S + kk0) * DM + h * DH + dh8;
        float4 x0 = *(const float4*)p, x1 = *(const float4*)(p + 4);
        r0[0] = x0.x; r0[1] = x0.y; r0[2] = x0.z; r0[3] = x0.w;
        r0[4] = x1.x; r0[5] = x1.y; r0[6] = x1.z; r0[7] = x1.w;
      }
      if (kk1 < S) {
        const float* p = Vm + (size_t)(b * S + kk1) * DM + h * DH + dh8;
        float4 x0 = *(const float4*)p, x1 = *(const float4*)(p + 4);
        r1[0] = x0.x; r1[1] = x0.y; r1[2] = x0.z; r1[3] = x0.w;
        r1[4] = x1.x; r1[5] = x1.y; r1[6] = x1.z; r1[7] = x1.w;
      }
#pragma unroll
      for (int j = 0; j < 8; ++j) {
        bf16x2 t;
        t[0] = (bf16_t)r0[j];
        t[1] = (bf16_t)r1[j];
        *(bf16x2*)&sVt[(dh8 + j) * LT + 2 * kp] = t;
      }
    }
    __syncthreads();

    {  // S tile = (Q @ K^T) * 1/sqrt(DH); each wave 2 of 16 16x16 tiles
      v8f sacc[2] = {};
#pragma unroll
      for (int ks = 0; ks < 2; ++ks) {
        v16bf af = load_afrag(sQ + (wm * 16 + l16) * LT + ks * 32, hl);
#pragma unroll
        for (int ni = 0; ni < 2; ++ni) {
          v16bf bfg =
              load_bfrag(sK + ((wn2 * 2 + ni) * 16 + l16) * LT + ks * 32, hl);
          sacc[ni] = wmma_bf16(af, bfg, sacc[ni]);
        }
      }
#pragma unroll
      for (int ni = 0; ni < 2; ++ni) {
        int col = (wn2 * 2 + ni) * 16 + l16;
        int rb = wm * 16 + hl * 8;
#pragma unroll
        for (int v = 0; v < 8; ++v) sS[(rb + v) * LS + col] = sacc[ni][v] * 0.125f;
      }
    }
    __syncthreads();

    {  // online softmax pass 1: per-row tile max (4 threads/row)
      int r = tid >> 2, seg = (tid & 3) * 16;
      float mx = -1e30f;
      for (int c = 0; c < 16; ++c) {
        int cg = k0t + seg + c;
        float v = (cg < S) ? sS[r * LS + seg + c] : -1e30f;
        mx = fmaxf(mx, v);
      }
      red[r * 4 + (tid & 3)] = mx;
    }
    __syncthreads();
    if (tid < 64) {
      float mx = fmaxf(fmaxf(red[tid * 4], red[tid * 4 + 1]),
                       fmaxf(red[tid * 4 + 2], red[tid * 4 + 3]));
      float mnew = fmaxf(sMx[tid], mx);
      sAl[tid] = __expf(sMx[tid] - mnew);
      sMx[tid] = mnew;
    }
    __syncthreads();
    {  // pass 2: P = exp(S - m), partial row sums
      int r = tid >> 2, seg = (tid & 3) * 16;
      float m = sMx[r], sum = 0.f;
      for (int c = 0; c < 16; ++c) {
        int cg = k0t + seg + c;
        float p = (cg < S) ? __expf(sS[r * LS + seg + c] - m) : 0.f;
        sP[r * LT + seg + c] = (bf16_t)p;
        sum += p;
      }
      red[r * 4 + (tid & 3)] = sum;
    }
    __syncthreads();
    if (tid < 64)
      sL[tid] = sAl[tid] * sL[tid] + red[tid * 4] + red[tid * 4 + 1] +
                red[tid * 4 + 2] + red[tid * 4 + 3];

    {  // O = O*alpha + P @ V
      int rb = wm * 16 + hl * 8;
      float al[8];
#pragma unroll
      for (int v = 0; v < 8; ++v) al[v] = sAl[rb + v];
#pragma unroll
      for (int ni = 0; ni < 2; ++ni)
#pragma unroll
        for (int v = 0; v < 8; ++v) oacc[ni][v] *= al[v];
#pragma unroll
      for (int ks = 0; ks < 2; ++ks) {
        v16bf af = load_afrag(sP + (wm * 16 + l16) * LT + ks * 32, hl);
#pragma unroll
        for (int ni = 0; ni < 2; ++ni) {
          v16bf bfg =
              load_bfrag(sVt + ((wn2 * 2 + ni) * 16 + l16) * LT + ks * 32, hl);
          oacc[ni] = wmma_bf16(af, bfg, oacc[ni]);
        }
      }
    }
    __syncthreads();
  }

  {  // write O / l
    int rb = wm * 16 + hl * 8;
    float linv[8];
#pragma unroll
    for (int v = 0; v < 8; ++v) {
      float l = sL[rb + v];
      linv[v] = (l > 0.f) ? 1.f / l : 0.f;
    }
#pragma unroll
    for (int ni = 0; ni < 2; ++ni) {
      int dh = (wn2 * 2 + ni) * 16 + l16;
#pragma unroll
      for (int v = 0; v < 8; ++v) {
        int q = q0 + rb + v;
        if (q < S)
          Om[(size_t)(b * S + q) * DM + h * DH + dh] = oacc[ni][v] * linv[v];
      }
    }
  }
}

// ------------------------------------------------ foreign-key sequential MLP

__device__ void fk_step(float* eb, int src_t, int dst_t,
                        const float* __restrict__ W,
                        const float* __restrict__ bias, float* sx) {
  int tid = threadIdx.x;
  for (int i = tid; i < 3 * DM; i += 256)
    sx[i] = eb[(size_t)src_t * 3 * DM + i];
  __syncthreads();
  float acc[3][4];
#pragma unroll
  for (int r = 0; r < 3; ++r)
#pragma unroll
    for (int cc = 0; cc < 4; ++cc) acc[r][cc] = bias[tid + cc * 256];
  for (int k = 0; k < DM; ++k) {
    float x0 = sx[k], x1 = sx[DM + k], x2 = sx[2 * DM + k];
    const float* wr = W + (size_t)k * DM + tid;
#pragma unroll
    for (int cc = 0; cc < 4; ++cc) {
      float w = wr[cc * 256];
      acc[0][cc] = fmaf(x0, w, acc[0][cc]);
      acc[1][cc] = fmaf(x1, w, acc[1][cc]);
      acc[2][cc] = fmaf(x2, w, acc[2][cc]);
    }
  }
#pragma unroll
  for (int r = 0; r < 3; ++r)
#pragma unroll
    for (int cc = 0; cc < 4; ++cc)
      eb[((size_t)dst_t * 3 + r) * DM + tid + cc * 256] += tanhf(acc[r][cc]);
  __syncthreads();
}

__global__ __launch_bounds__(256) void fk_kernel(
    const float* __restrict__ tc, const int* __restrict__ parent_nums,
    const int* __restrict__ fkeys, const float* __restrict__ prim_w,
    const float* __restrict__ prim_b, const float* __restrict__ for_w,
    const float* __restrict__ for_b, float* __restrict__ enc) {
  __shared__ float sx[3 * DM];
  int b = blockIdx.x, tid = threadIdx.x;
  float* eb = enc + (size_t)b * (3 * NT) * DM;
  // gather first 3 tokens of each table-col sequence: enc[b, t*3+j] = tc[b*NT+t, j]
  for (int i = 0; i < 3 * NT; ++i) {
    const float* src = tc + ((size_t)(b * NT + i / 3) * NC + (i % 3)) * DM;
    *(float4*)(eb + (size_t)i * DM + tid * 4) = *(const float4*)(src + tid * 4);
  }
  __syncthreads();
  for (int e = 0; e < NFK; ++e) {
    int f = parent_nums[b * NP + fkeys[(b * NFK + e) * 2 + 0]];
    int p = parent_nums[b * NP + fkeys[(b * NFK + e) * 2 + 1]];
    fk_step(eb, p, f, prim_w, prim_b, sx);  // tab[f] += tanh(tab[p]@prim_w+b)
    fk_step(eb, f, p, for_w, for_b, sx);    // tab[p] += tanh(tab[f]@for_w+b)
  }
}

// ---------------------------------------------------------------- host side

struct LayerW {
  const float *qkv_b, *ob, *ln1, *b1, *b2, *ln2;
  const bf16_t *qkvT, *oT, *w1T, *w2T;  // pre-transposed bf16 weights [N,K]
};

static void run_bert_layer(const LayerW& W, int l, float* x, int M, int Batt,
                           int S, float* Qb, float* Kb, float* Vb, float* Tb,
                           float* Hb, hipStream_t stream) {
  dim3 blk(256);
  dim3 gD(DM / 128, M / 128);
  const size_t dd = (size_t)DM * DM;
  float* qkv[3] = {Qb, Kb, Vb};
  for (int m = 0; m < 3; ++m)
    gemm_bf16_kernel<0><<<gD, blk, 0, stream>>>(
        x, W.qkvT + (size_t)(l * 3 + m) * dd, W.qkv_b + (size_t)(l * 3 + m) * DM,
        nullptr, qkv[m], M, DM, DM);
  attn_kernel<<<dim3(Batt * NH, (S + 63) / 64), blk, 0, stream>>>(Qb, Kb, Vb, Tb,
                                                                  S);
  gemm_bf16_kernel<0><<<gD, blk, 0, stream>>>(Tb, W.oT + (size_t)l * dd,
                                              W.ob + (size_t)l * DM, x, Qb, M,
                                              DM, DM);
  ln_kernel<<<dim3(M), blk, 0, stream>>>(Qb, W.ln1 + (size_t)l * 2 * DM,
                                         W.ln1 + (size_t)l * 2 * DM + DM, x);
  gemm_bf16_kernel<1><<<dim3(DFF / 128, M / 128), blk, 0, stream>>>(
      x, W.w1T + (size_t)l * DM * DFF, W.b1 + (size_t)l * DFF, nullptr, Hb, M,
      DFF, DM);
  gemm_bf16_kernel<0><<<gD, blk, 0, stream>>>(Hb, W.w2T + (size_t)l * DFF * DM,
                                              W.b2 + (size_t)l * DM, x, Qb, M,
                                              DM, DFF);
  ln_kernel<<<dim3(M), blk, 0, stream>>>(Qb, W.ln2 + (size_t)l * 2 * DM,
                                         W.ln2 + (size_t)l * 2 * DM + DM, x);
}

extern "C" void kernel_launch(void* const* d_in, const int* in_sizes, int n_in,
                              void* d_out, int out_size, void* d_ws,
                              size_t ws_size, hipStream_t stream) {
  (void)in_sizes; (void)n_in; (void)out_size; (void)ws_size;
  const int* input_ids = (const int*)d_in[0];
  const int* table_cols = (const int*)d_in[1];
  const int* table_types = (const int*)d_in[2];
  const int* parent_nums = (const int*)d_in[3];
  const int* fkeys = (const int*)d_in[4];
  const float* word_emb = (const float*)d_in[9];
  const float* pos_emb = (const float*)d_in[10];
  const float* type_emb = (const float*)d_in[11];
  const float* emb_ln = (const float*)d_in[12];
  const float* qkv_w = (const float*)d_in[13];
  const float* attn_o_w = (const float*)d_in[15];
  const float* ffn_w1 = (const float*)d_in[18];
  const float* ffn_w2 = (const float*)d_in[20];
  LayerW W;
  W.qkv_b = (const float*)d_in[14];
  W.ob = (const float*)d_in[16];
  W.ln1 = (const float*)d_in[17];
  W.b1 = (const float*)d_in[19];
  W.b2 = (const float*)d_in[21];
  W.ln2 = (const float*)d_in[22];
  const float* prim_w = (const float*)d_in[23];
  const float* prim_b = (const float*)d_in[24];
  const float* for_w = (const float*)d_in[25];
  const float* for_b = (const float*)d_in[26];

  const int M1 = NB * NT * NC;  // 15360 table tokens (layer 0)
  const int M2 = NB * SM;       // 12032 main tokens (layers 1..4)
  float* X = (float*)d_ws;                 // [M1, DM]
  float* Qb = X + (size_t)M1 * DM;         // [M1, DM]
  float* Kb = Qb + (size_t)M1 * DM;        // [M1, DM]
  float* Vb = Kb + (size_t)M1 * DM;        // [M1, DM]
  float* Tb = Vb + (size_t)M1 * DM;        // [M1, DM]
  float* Hb = Tb + (size_t)M1 * DM;        // [M1, DFF]
  float* ENC = Hb + (size_t)M1 * DFF;      // [NB, 3*NT, DM]
  bf16_t* qkvT = (bf16_t*)(ENC + (size_t)NB * 3 * NT * DM);  // 15x[DM,DM]
  bf16_t* oT = qkvT + (size_t)NLAY * 3 * DM * DM;            // 5x[DM,DM]
  bf16_t* w1T = oT + (size_t)NLAY * DM * DM;                 // 5x[DFF,DM]
  bf16_t* w2T = w1T + (size_t)NLAY * DM * DFF;               // 5x[DM,DFF]
  W.qkvT = qkvT; W.oT = oT; W.w1T = w1T; W.w2T = w2T;
  float* xmain = (float*)d_out;            // [NB, SM, DM]

  dim3 blk(256);
  // one-time (per launch) weight fp32[K,N] -> bf16[N,K] convert+transpose
  transpose_cvt_kernel<<<dim3(DM / 32, DM / 32, NLAY * 3), blk, 0, stream>>>(
      qkv_w, qkvT, DM, DM);
  transpose_cvt_kernel<<<dim3(DM / 32, DM / 32, NLAY), blk, 0, stream>>>(
      attn_o_w, oT, DM, DM);
  transpose_cvt_kernel<<<dim3(DFF / 32, DM / 32, NLAY), blk, 0, stream>>>(
      ffn_w1, w1T, DM, DFF);
  transpose_cvt_kernel<<<dim3(DM / 32, DFF / 32, NLAY), blk, 0, stream>>>(
      ffn_w2, w2T, DFF, DM);

  // question embedding -> rows [b, 0..255] of output
  embed_ln_kernel<<<dim3(NB * LQ), blk, 0, stream>>>(
      input_ids, nullptr, word_emb, pos_emb, type_emb, emb_ln, xmain, LQ, SM);
  // table-col embedding -> X
  embed_ln_kernel<<<dim3(M1), blk, 0, stream>>>(table_cols, table_types,
                                                word_emb, pos_emb, type_emb,
                                                emb_ln, X, NC, NC);
  // table-cols encoder (layer 0): 1280 sequences of length 12
  run_bert_layer(W, 0, X, M1, NB * NT, NC, Qb, Kb, Vb, Tb, Hb, stream);
  // foreign-key gather/scatter MLP scan on first-3-token slices
  fk_kernel<<<dim3(NB), blk, 0, stream>>>(X, parent_nums, fkeys, prim_w, prim_b,
                                          for_w, for_b, ENC);
  // assemble table rows: +pos/type emb, LN -> rows [b, 256..375] of output
  addposln_kernel<<<dim3(NB * 3 * NT), blk, 0, stream>>>(ENC, pos_emb, type_emb,
                                                         emb_ln, xmain);
  // main encoder layers 1..4 on [NB, 376, DM]
  for (int l = 1; l < NLAY; ++l)
    run_bert_layer(W, l, xmain, M2, NB, SM, Qb, Kb, Vb, Tb, Hb, stream);
}
